// EncoderRNN_9440338117097
// MI455X (gfx1250) — compile-verified, matrix operand order checked by reading, not run on hardware
//
#include <hip/hip_runtime.h>
#include <hip/hip_bf16.h>

// ---------------------------------------------------------------------------
// GRU encoder for MI455X (gfx1250), wave32 + WMMA bf16.
//   VOCAB=32000, EMB=HID=1024, B=128, T=512, 3H=3072
// Pipeline:
//   1) pack weights (fp32 -> bf16, WMMA B-fragment swizzle)
//   2) gather+pack embeddings into WMMA A-fragment swizzle (x, time-major)
//   3) big GEMM: gi[t*128+b][3072] = x . W_ih^T + b_ih   (bf16 out)
//   4) persistent scan kernel: 512 steps, grid sync per step,
//      gh = h . W_hh^T via v_wmma_f32_16x16x32_bf16, fused gates,
//      h kept in VGPRs per wave, bf16 packed h ping-pong for GEMM reads,
//      per-WG W_hh slice (96 KB, step-invariant) cached in LDS for the
//      entire scan -> hot loop reads B fragments via ds_load_b128.
// Hot loops are explicitly double-buffered so loads overlap the WMMA pipe.
// ---------------------------------------------------------------------------

typedef __bf16 bf16x16 __attribute__((ext_vector_type(16)));
typedef float  f32x8   __attribute__((ext_vector_type(8)));

#define N_EMB   1024
#define N_HID   1024
#define N_G3    3072
#define N_BATCH 128
#define N_SEQ   512
#define N_ROWS  (N_SEQ * N_BATCH)           // 65536 rows of x / gi

// tile geometry: 16x32 (A) and 32x16 (B) bf16 fragments = 512 elems = 1KB
#define KT_PER_K  (N_EMB / 32)              // 32 k-tiles
#define TILE_ELEMS 512
#define SCAN_LDS_ELEMS (3 * KT_PER_K * TILE_ELEMS)   // 49152 bf16 = 96 KB

__device__ __forceinline__ f32x8 wmma_bf16(bf16x16 a, bf16x16 b, f32x8 c) {
    return __builtin_amdgcn_wmma_f32_16x16x32_bf16(
        false, a, false, b, (short)0, c, false, false);
}

// CDNA5 16-bit A/B fragment swizzle (ISA 7.12.2):
// lane l holds 16 elems; element e of lane l corresponds to local
//   k = 16*(e>=8) + 8*(l>=16) + (e&7),  row/col = l&15.
// Inverse: given (rc in 0..15, kl in 0..31) -> in-tile offset l*16+e.
__device__ __forceinline__ int frag_off(int rc, int kl) {
    int l = rc + ((kl & 8) ? 16 : 0);
    int e = ((kl & 16) ? 8 : 0) + (kl & 7);
    return l * 16 + e;
}

__device__ __forceinline__ float sigmoidf_fast(float x) {
    return 1.0f / (1.0f + __expf(-x));
}

// ---------------------------------------------------------------------------
// Kernel 1: pack a [3072,1024] fp32 weight into bf16 B-fragments
// ---------------------------------------------------------------------------
__global__ __launch_bounds__(256) void pack_w_kernel(const float* __restrict__ w,
                                                     __bf16* __restrict__ dst) {
    size_t tid = (size_t)blockIdx.x * 256 + threadIdx.x;   // 3072*1024 threads
    int n = (int)(tid >> 10);
    int k = (int)(tid & 1023);
    float v = w[tid];
    size_t off = ((size_t)(n >> 4) * KT_PER_K + (k >> 5)) * TILE_ELEMS
               + frag_off(n & 15, k & 31);
    dst[off] = (__bf16)v;
}

// ---------------------------------------------------------------------------
// Kernel 2: embedding gather + pack into A-fragments, time-major rows.
//   row = t*128 + b ; x[row][k] = emb[seq[b*512+t]][k]
// ---------------------------------------------------------------------------
__global__ __launch_bounds__(256) void gather_pack_x_kernel(
        const int* __restrict__ seq, const float* __restrict__ emb,
        __bf16* __restrict__ xp) {
    size_t tid = (size_t)blockIdx.x * 256 + threadIdx.x;   // 65536*1024 threads
    int row = (int)(tid >> 10);
    int k   = (int)(tid & 1023);
    int t = row >> 7;
    int b = row & 127;
    int tok = seq[b * N_SEQ + t];
    float v = emb[(size_t)tok * N_EMB + k];
    size_t off = ((size_t)(row >> 4) * KT_PER_K + (k >> 5)) * TILE_ELEMS
               + frag_off(row & 15, k & 31);
    xp[off] = (__bf16)v;
}

// ---------------------------------------------------------------------------
// Kernel 3: gi = x . W_ih^T + b_ih   -> bf16 [65536][3072]
// Grid: (512, 24) WGs of 256 threads (8 waves).
// Wave w: m-tile = bx*8+w ; covers n-tiles by*8 .. by*8+7 (128 cols).
// All 8 waves of a WG stream identical B fragments -> WGP$ hits.
// Double-buffered k-loop: next k-step's loads issued before current WMMAs.
// ---------------------------------------------------------------------------
__global__ __launch_bounds__(256) void gemm_gi_kernel(
        const __bf16* __restrict__ xp, const __bf16* __restrict__ wih,
        const float* __restrict__ b_ih, __bf16* __restrict__ gi) {
    const int lane = threadIdx.x & 31;
    const int wave = threadIdx.x >> 5;
    const int mt   = blockIdx.x * 8 + wave;       // 0..4095
    const int ntB  = blockIdx.y * 8;              // n-tile base (of 192)

    f32x8 zero = {0.f,0.f,0.f,0.f,0.f,0.f,0.f,0.f};
    f32x8 acc[8];
#pragma unroll
    for (int j = 0; j < 8; ++j) acc[j] = zero;

    const __bf16* aBase = xp + (size_t)mt * KT_PER_K * TILE_ELEMS + lane * 16;
    const __bf16* bBase = wih + (size_t)ntB * KT_PER_K * TILE_ELEMS + lane * 16;

#define LOAD_FRAGS(KT, AV, BV)                                            \
    do {                                                                  \
        AV = *(const bf16x16*)(aBase + (size_t)(KT) * TILE_ELEMS);        \
        _Pragma("unroll")                                                 \
        for (int j = 0; j < 8; ++j)                                       \
            BV[j] = *(const bf16x16*)(bBase +                             \
                ((size_t)j * KT_PER_K + (KT)) * TILE_ELEMS);              \
    } while (0)

    bf16x16 aX, bX[8], aY, bY[8];
    LOAD_FRAGS(0, aX, bX);
    for (int kt = 0; kt < KT_PER_K; kt += 2) {
        LOAD_FRAGS(kt + 1, aY, bY);           // overlap with WMMAs below
        __builtin_prefetch(aBase + (size_t)(kt + 2) * TILE_ELEMS, 0, 1);
#pragma unroll
        for (int j = 0; j < 8; ++j) acc[j] = wmma_bf16(aX, bX[j], acc[j]);
        if (kt + 2 < KT_PER_K) LOAD_FRAGS(kt + 2, aX, bX);
#pragma unroll
        for (int j = 0; j < 8; ++j) acc[j] = wmma_bf16(aY, bY[j], acc[j]);
    }
#undef LOAD_FRAGS

    // epilogue: + b_ih, store bf16 row-major gi[row][n]
    const int half = lane >> 4;    // C layout: M = r + 8*half, N = lane&15
#pragma unroll
    for (int j = 0; j < 8; ++j) {
        int n = (ntB + j) * 16 + (lane & 15);
        float bias = b_ih[n];
#pragma unroll
        for (int r = 0; r < 8; ++r) {
            int row = mt * 16 + r + (half << 3);
            gi[(size_t)row * N_G3 + n] = (__bf16)(acc[j][r] + bias);
        }
    }
}

// ---------------------------------------------------------------------------
// Kernel 4: persistent GRU scan. 64 WGs x 256 thr = 512 waves.
// Mapping: jt = blockIdx.x (hidden tile 0..63), mt = wave (batch tile 0..7).
// The WG's W_hh slice (3 gates x 32 k-tiles x 1KB = 96 KB) is step-invariant:
// loaded ONCE into LDS before the time loop; all 512 steps x 8 waves read
// B fragments with ds_load_b128. Per-step global traffic: h ping-pong + gi.
// Each wave: gh r/z/n strips for its 16b x 16j tile, fused gates, h in VGPRs.
// One grid-wide sync per step (atomic counter + acquire spin).
// ---------------------------------------------------------------------------
__global__ __launch_bounds__(256) void gru_scan_kernel(
        const __bf16* __restrict__ gi, const __bf16* __restrict__ whh,
        const float* __restrict__ b_hh,
        __bf16* __restrict__ hbf0, __bf16* __restrict__ hbf1,
        float* __restrict__ out, unsigned* __restrict__ sync) {
    extern __shared__ __bf16 lds_w[];         // 49152 bf16 = 96 KB
    const int lane = threadIdx.x & 31;
    const int wave = threadIdx.x >> 5;
    const int jt = blockIdx.x;                // hidden tile 0..63 (shared in WG)
    const int mt = wave;                      // batch tile 0..7
    const int j  = (jt << 4) + (lane & 15);   // hidden col 0..1023
    const int half = lane >> 4;

    // ---- one-time cooperative LDS fill of the WG's W_hh slice ----
    // gate g lives at lds_w[g*32*512 ..), tiles (kt) contiguous, frag-swizzled.
    {
        uint4* dst = (uint4*)lds_w;           // 6144 x 16B total
#pragma unroll
        for (int g = 0; g < 3; ++g) {
            const uint4* src = (const uint4*)(whh +
                ((size_t)(g * 64 + jt) * KT_PER_K) * TILE_ELEMS);
#pragma unroll
            for (int i = 0; i < 8; ++i) {     // 2048 chunks per gate
                int c = i * 256 + threadIdx.x;
                dst[g * 2048 + c] = src[c];
            }
        }
        __syncthreads();
    }

    const float bhr = b_hh[j];
    const float bhz = b_hh[N_HID + j];
    const float bhn = b_hh[2 * N_HID + j];

    // LDS B-fragment bases for the 3 gates
    const __bf16* lr = lds_w + lane * 16;
    const __bf16* lz = lds_w + (size_t)KT_PER_K * TILE_ELEMS + lane * 16;
    const __bf16* ln = lds_w + (size_t)2 * KT_PER_K * TILE_ELEMS + lane * 16;

    // packed-h write offsets (j fixed per lane)
    const int kl  = j & 31;
    const int ktw = j >> 5;
    const int lhi = (kl & 8) ? 16 : 0;
    const int e   = ((kl & 16) ? 8 : 0) + (kl & 7);
    const size_t wbase = ((size_t)mt * KT_PER_K + ktw) * TILE_ELEMS + e;

    float hprev[8];
#pragma unroll
    for (int r = 0; r < 8; ++r) hprev[r] = 0.0f;

    f32x8 zero = {0.f,0.f,0.f,0.f,0.f,0.f,0.f,0.f};

    for (int t = 0; t < N_SEQ; ++t) {
        const __bf16* hread = (t & 1) ? hbf1 : hbf0;
        __bf16*       hwrit = (t & 1) ? hbf0 : hbf1;

        f32x8 accR = zero, accZ = zero, accN = zero;
        const __bf16* ha = hread + (size_t)mt * KT_PER_K * TILE_ELEMS + lane * 16;

#define LOAD_SFRAGS(KT, AV, RV, ZV, NV)                                   \
    do {                                                                  \
        AV = *(const bf16x16*)(ha + (size_t)(KT) * TILE_ELEMS);           \
        RV = *(const bf16x16*)(lr + (size_t)(KT) * TILE_ELEMS);           \
        ZV = *(const bf16x16*)(lz + (size_t)(KT) * TILE_ELEMS);           \
        NV = *(const bf16x16*)(ln + (size_t)(KT) * TILE_ELEMS);           \
    } while (0)

        bf16x16 aX, rX, zX, nX, aY, rY, zY, nY;
        LOAD_SFRAGS(0, aX, rX, zX, nX);
        for (int kt = 0; kt < KT_PER_K; kt += 2) {
            LOAD_SFRAGS(kt + 1, aY, rY, zY, nY);   // overlap with WMMAs
            accR = wmma_bf16(aX, rX, accR);
            accZ = wmma_bf16(aX, zX, accZ);
            accN = wmma_bf16(aX, nX, accN);
            if (kt + 2 < KT_PER_K) LOAD_SFRAGS(kt + 2, aX, rX, zX, nX);
            accR = wmma_bf16(aY, rY, accR);
            accZ = wmma_bf16(aY, zY, accZ);
            accN = wmma_bf16(aY, nY, accN);
        }
#undef LOAD_SFRAGS

        // fused gates for this wave's 16x16 h tile (8 elems / lane)
        const __bf16* grow = gi + ((size_t)t * N_BATCH + mt * 16) * N_G3;
        if (t + 1 < N_SEQ)
            __builtin_prefetch(grow + (size_t)N_BATCH * N_G3, 0, 1);
#pragma unroll
        for (int r = 0; r < 8; ++r) {
            int m = r + (half << 3);                       // local batch row
            const __bf16* g = grow + (size_t)m * N_G3;
            float gir = (float)g[j];
            float giz = (float)g[N_HID + j];
            float gin = (float)g[2 * N_HID + j];
            float rr = sigmoidf_fast(gir + accR[r] + bhr);
            float zz = sigmoidf_fast(giz + accZ[r] + bhz);
            float nn = tanhf(gin + rr * (accN[r] + bhn));
            float hn = (1.0f - zz) * nn + zz * hprev[r];
            hprev[r] = hn;
            hwrit[wbase + (size_t)(m + lhi) * 16] = (__bf16)hn;
        }

        // ---- grid-wide step barrier ----
        __threadfence();
        __syncthreads();
        if (threadIdx.x == 0) {
            __hip_atomic_fetch_add(sync, 1u, __ATOMIC_RELEASE,
                                   __HIP_MEMORY_SCOPE_AGENT);
            unsigned target = (unsigned)(t + 1) * gridDim.x;
            while (__hip_atomic_load(sync, __ATOMIC_ACQUIRE,
                                     __HIP_MEMORY_SCOPE_AGENT) < target)
                __builtin_amdgcn_s_sleep(2);
        }
        __syncthreads();
    }

    // final hidden state -> d_out [1,128,1024] fp32
#pragma unroll
    for (int r = 0; r < 8; ++r) {
        int b = mt * 16 + r + (half << 3);
        out[(size_t)b * N_HID + j] = hprev[r];
    }
}

// ---------------------------------------------------------------------------
// Host launcher
// ---------------------------------------------------------------------------
extern "C" void kernel_launch(void* const* d_in, const int* in_sizes, int n_in,
                              void* d_out, int out_size, void* d_ws, size_t ws_size,
                              hipStream_t stream) {
    (void)in_sizes; (void)n_in; (void)out_size; (void)ws_size;
    const int*   seq  = (const int*)  d_in[0];
    const float* emb  = (const float*)d_in[1];
    const float* wih  = (const float*)d_in[2];
    const float* whh  = (const float*)d_in[3];
    const float* b_ih = (const float*)d_in[4];
    const float* b_hh = (const float*)d_in[5];
    float* out = (float*)d_out;

    // workspace layout (bytes, 256-aligned)
    char* ws = (char*)d_ws;
    size_t off = 0;
    auto alloc = [&](size_t bytes) {
        char* p = ws + off;
        off = (off + bytes + 255) & ~(size_t)255;
        return p;
    };
    unsigned* syncp   = (unsigned*)alloc(256);
    __bf16* wih_p     = (__bf16*)alloc((size_t)N_G3 * N_EMB * 2);      // 6.3 MB
    __bf16* whh_p     = (__bf16*)alloc((size_t)N_G3 * N_HID * 2);      // 6.3 MB
    __bf16* x_p       = (__bf16*)alloc((size_t)N_ROWS * N_EMB * 2);    // 134 MB
    __bf16* gi_p      = (__bf16*)alloc((size_t)N_ROWS * N_G3 * 2);     // 403 MB
    __bf16* hbf0      = (__bf16*)alloc((size_t)N_BATCH * N_HID * 2);   // 256 KB
    __bf16* hbf1      = (__bf16*)alloc((size_t)N_BATCH * N_HID * 2);   // 256 KB

    hipMemsetAsync(syncp, 0, 256, stream);
    hipMemsetAsync(hbf0, 0, (size_t)N_BATCH * N_HID * 2, stream);
    hipMemsetAsync(hbf1, 0, (size_t)N_BATCH * N_HID * 2, stream);

    // 1) pack weights
    {
        size_t n = (size_t)N_G3 * N_EMB;                  // 3.1M
        pack_w_kernel<<<(unsigned)(n / 256), 256, 0, stream>>>(wih, wih_p);
        pack_w_kernel<<<(unsigned)(n / 256), 256, 0, stream>>>(whh, whh_p);
    }
    // 2) gather + pack x
    {
        size_t n = (size_t)N_ROWS * N_EMB;                // 67.1M
        gather_pack_x_kernel<<<(unsigned)(n / 256), 256, 0, stream>>>(seq, emb, x_p);
    }
    // 3) gi GEMM: 65536 x 3072 x 1024
    {
        dim3 grid(N_ROWS / 128, N_G3 / 128);              // (512, 24)
        gemm_gi_kernel<<<grid, 256, 0, stream>>>(x_p, wih_p, b_ih, gi_p);
    }
    // 4) persistent recurrent scan (96 KB dynamic LDS for the W_hh slice)
    gru_scan_kernel<<<64, 256, SCAN_LDS_ELEMS * sizeof(__bf16), stream>>>(
        gi_p, whh_p, b_hh, hbf0, hbf1, out, syncp);
}
